// Conv_MS_MSA_90520730730927
// MI455X (gfx1250) — compile-verified, matrix operand order checked by reading, not run on hardware
//
#include <hip/hip_runtime.h>
#include <hip/hip_bf16.h>

#define BATCH 2
#define HH 256
#define WW 256
#define CC 256
#define NN (HH * WW)      // 65536
#define CQKV 768
#define NHEADS 8
#define DHEAD 32

typedef __bf16 bf16_t;
typedef __attribute__((ext_vector_type(16))) __bf16 v16bf;
typedef __attribute__((ext_vector_type(8)))  __bf16 v8bf;
typedef __attribute__((ext_vector_type(8)))  float  v8f;

// Native converts: backend emits hardware bf16 cvt ops (no manual RNE bit math).
__device__ __forceinline__ bf16_t f2bf(float f) { return (bf16_t)f; }
__device__ __forceinline__ float  bf2f(bf16_t b) { return (float)b; }

// K-permutation: swap the two middle groups of 8 (bit3 <-> bit4), so each
// lane's 16 fragment elements become one contiguous 32B run in LDS.
__device__ __forceinline__ int permk(int k) {
    return (k & ~24) | ((k & 8) << 1) | ((k & 16) >> 1);
}

// Fragment from a permuted LDS tile: lane reads 16 contiguous bf16
// (two ds_load_b128). stride in elements (multiple of 8 keeps 16B alignment).
__device__ __forceinline__ v16bf frag_lds(const bf16_t* tile, int stride, int lane) {
    const bf16_t* p = tile + (lane & 15) * stride + ((lane >> 4) << 4);
    union { v16bf v; v8bf h[2]; } u;
    u.h[0] = *(const v8bf*)p;
    u.h[1] = *(const v8bf*)(p + 8);
    return u.v;
}

// Fragment straight from global memory when the K-runs are contiguous:
// p points at row + kbase; elements [0..7] and [16..23] (two global_load_b128).
__device__ __forceinline__ v16bf frag_glb(const bf16_t* p) {
    union { v16bf v; v8bf h[2]; } u;
    u.h[0] = *(const v8bf*)p;
    u.h[1] = *(const v8bf*)(p + 16);
    return u.v;
}

#define WMMA_BF16(a, b, c) \
    __builtin_amdgcn_wmma_f32_16x16x32_bf16(false, (a), false, (b), (short)0, (c), false, false)

#define XSTR 40   // padded tile stride (multiple of 8 for alignment, dodges banks)

// =====================================================================
// k1: 1x1 QKV conv == GEMM [B*N,256] x [256,768] -> qkv bf16 (b,c,n) layout
// =====================================================================
__global__ __launch_bounds__(256) void k1_qkv(const float* __restrict__ x,
                                              const float* __restrict__ wqkv,
                                              bf16_t* __restrict__ qkv) {
    __shared__ bf16_t Xs[64 * XSTR];     // [m 64][k 32 permuted]
    __shared__ bf16_t Ws[64 * XSTR];     // [co 64][k 32 permuted]
    __shared__ bf16_t Cs[64 * 72];       // [co 64][m 64] staging for stores
    int m0 = blockIdx.x * 64;            // over flattened b*N
    int n0 = blockIdx.y * 64;            // over 768 output channels
    int tid = threadIdx.x, wave = tid >> 5, lane = tid & 31;
    int mt = wave & 3, ntp = wave >> 2;
    v8f acc0 = {}, acc1 = {};
    for (int ci0 = 0; ci0 < CC; ci0 += 32) {
        {   // X tile: each thread converts 8 contiguous floats -> one b128 store
            int e0 = tid * 8;
            int row = e0 >> 5, col = e0 & 31;           // col in {0,8,16,24}
            const float* src = x + (size_t)(m0 + row) * CC + ci0 + col;
            float4 f0 = *(const float4*)src;
            float4 f1 = *(const float4*)(src + 4);
            v8bf t;
            t[0] = f2bf(f0.x); t[1] = f2bf(f0.y); t[2] = f2bf(f0.z); t[3] = f2bf(f0.w);
            t[4] = f2bf(f1.x); t[5] = f2bf(f1.y); t[6] = f2bf(f1.z); t[7] = f2bf(f1.w);
            *(v8bf*)&Xs[row * XSTR + permk(col)] = t;
        }
        {   // W tile
            int e0 = tid * 8;
            int co = e0 >> 5, k0 = e0 & 31;
            const float* src = wqkv + (size_t)(n0 + co) * CC + ci0 + k0;
            float4 f0 = *(const float4*)src;
            float4 f1 = *(const float4*)(src + 4);
            v8bf t;
            t[0] = f2bf(f0.x); t[1] = f2bf(f0.y); t[2] = f2bf(f0.z); t[3] = f2bf(f0.w);
            t[4] = f2bf(f1.x); t[5] = f2bf(f1.y); t[6] = f2bf(f1.z); t[7] = f2bf(f1.w);
            *(v8bf*)&Ws[co * XSTR + permk(k0)] = t;
        }
        __syncthreads();
        v16bf a  = frag_lds(&Xs[(mt * 16) * XSTR], XSTR, lane);
        v16bf b0 = frag_lds(&Ws[(ntp * 32) * XSTR], XSTR, lane);
        v16bf b1 = frag_lds(&Ws[(ntp * 32 + 16) * XSTR], XSTR, lane);
        acc0 = WMMA_BF16(a, b0, acc0);
        acc1 = WMMA_BF16(a, b1, acc1);
        __syncthreads();
    }
    // stage C tile in LDS, then write contiguous 128-bit runs along n
    int lhalf = lane >> 4, nlane = lane & 15;
#pragma unroll
    for (int r = 0; r < 8; ++r) {
        int ml = mt * 16 + r + lhalf * 8;
        Cs[(ntp * 32 + nlane) * 72 + ml]      = f2bf(acc0[r]);
        Cs[(ntp * 32 + nlane + 16) * 72 + ml] = f2bf(acc1[r]);
    }
    __syncthreads();
    int row = tid >> 2;                  // co_local 0..63
    int mo  = (tid & 3) * 16;            // 16 m's per thread
    int b = m0 / NN, nb = m0 % NN;
    bf16_t* dst = qkv + ((size_t)b * CQKV + n0 + row) * NN + nb + mo;
    *(v8bf*)dst       = *(const v8bf*)&Cs[row * 72 + mo];
    *(v8bf*)(dst + 8) = *(const v8bf*)&Cs[row * 72 + mo + 8];
}

// =====================================================================
// k2: depthwise 3x3 (groups=768); split into q,k (bf16) + v_out (f32),
//     plus per-(b,channel) sum-of-squares for q,k (norm over n).
// =====================================================================
__global__ __launch_bounds__(256) void k2_dw(const bf16_t* __restrict__ qkv,
                                             const float* __restrict__ wdw,
                                             bf16_t* __restrict__ qb,
                                             bf16_t* __restrict__ kb,
                                             float* __restrict__ vout,
                                             float* __restrict__ ssq,
                                             float* __restrict__ ssk) {
    const int chunks = NN / 256;
    int bx = blockIdx.x;
    int chunk = bx % chunks;
    int c = (bx / chunks) % CQKV;
    int b = bx / (chunks * CQKV);
    int tid = threadIdx.x;
    int n = chunk * 256 + tid;
    int y = n >> 8, x = n & 255;
    const bf16_t* base = qkv + ((size_t)b * CQKV + c) * NN;
    float wv[9];
#pragma unroll
    for (int i = 0; i < 9; ++i) wv[i] = wdw[c * 9 + i];
    float acc = 0.f;
#pragma unroll
    for (int ky = -1; ky <= 1; ++ky) {
        int yy = y + ky;
        if (yy < 0 || yy >= HH) continue;
#pragma unroll
        for (int kx = -1; kx <= 1; ++kx) {
            int xx = x + kx;
            if (xx < 0 || xx >= WW) continue;
            acc += bf2f(base[yy * WW + xx]) * wv[(ky + 1) * 3 + (kx + 1)];
        }
    }
    __shared__ float red[256];
    if (c < CC) {
        qb[((size_t)b * CC + c) * NN + n] = f2bf(acc);
        red[tid] = acc * acc;
    } else if (c < 2 * CC) {
        kb[((size_t)b * CC + (c - CC)) * NN + n] = f2bf(acc);
        red[tid] = acc * acc;
    } else {
        vout[((size_t)b * CC + (c - 2 * CC)) * NN + n] = acc;  // v_out (NCHW), f32
    }
    if (c < 2 * CC) {              // uniform per block
        __syncthreads();
        for (int s = 128; s > 0; s >>= 1) {
            if (tid < s) red[tid] += red[tid + s];
            __syncthreads();
        }
        if (tid == 0) {
            if (c < CC) atomicAdd(&ssq[b * CC + c], red[0]);
            else        atomicAdd(&ssk[b * CC + (c - CC)], red[0]);
        }
    }
}

// =====================================================================
// k3: raw Gram G[b,h,d,e] = sum_n k[d,n]*q[e,n] via WMMA; fragments read
//     directly from global as two b128 loads each (rows contiguous in n).
// =====================================================================
__global__ __launch_bounds__(256) void k3_gram(const bf16_t* __restrict__ qb,
                                               const bf16_t* __restrict__ kb,
                                               float* __restrict__ gram) {
    const int CHUNK = 2048;
    const int NCH = NN / CHUNK;  // 32
    int bx = blockIdx.x;
    int chunk = bx % NCH;
    int head = (bx / NCH) % NHEADS;
    int b = bx / (NCH * NHEADS);
    int tid = threadIdx.x, wave = tid >> 5, lane = tid & 31;
    const bf16_t* kp = kb + ((size_t)b * CC + head * DHEAD) * NN;
    const bf16_t* qp = qb + ((size_t)b * CC + head * DHEAD) * NN;
    v8f acc[2][2] = {};
    int nwbase = chunk * CHUNK + wave * 256;
    int nlane = lane & 15;
    int kb8 = (lane >> 4) ? 8 : 0;
    for (int s = 0; s < 8; ++s) {
        int n0 = nwbase + s * 32;
        v16bf af[2], bfr[2];
#pragma unroll
        for (int t = 0; t < 2; ++t) {
            int de = t * 16 + nlane;
            af[t]  = frag_glb(kp + (size_t)de * NN + n0 + kb8);   // A[d, n]
            bfr[t] = frag_glb(qp + (size_t)de * NN + n0 + kb8);   // B[n, e]
        }
#pragma unroll
        for (int dt = 0; dt < 2; ++dt)
#pragma unroll
            for (int et = 0; et < 2; ++et)
                acc[dt][et] = WMMA_BF16(af[dt], bfr[et], acc[dt][et]);
    }
    __shared__ float gs[32 * 32];
    for (int e = tid; e < 1024; e += 256) gs[e] = 0.f;
    __syncthreads();
    int lhalf = lane >> 4;
#pragma unroll
    for (int dt = 0; dt < 2; ++dt)
#pragma unroll
        for (int et = 0; et < 2; ++et)
#pragma unroll
            for (int r = 0; r < 8; ++r) {
                int d = dt * 16 + r + lhalf * 8;
                int e = et * 16 + nlane;
                atomicAdd(&gs[d * 32 + e], acc[dt][et][r]);
            }
    __syncthreads();
    float* g = gram + (size_t)(b * NHEADS + head) * 1024;
    for (int e = tid; e < 1024; e += 256) atomicAdd(&g[e], gs[e]);
}

// =====================================================================
// k4: fold normalization + rescale, softmax over e. One wave per (b,head).
// =====================================================================
__global__ __launch_bounds__(32) void k4_soft(const float* __restrict__ gram,
                                              const float* __restrict__ ssq,
                                              const float* __restrict__ ssk,
                                              const float* __restrict__ rescale,
                                              float* __restrict__ A) {
    int bh = blockIdx.x;
    int head = bh % NHEADS, b = bh / NHEADS;
    int e = threadIdx.x;
    float qn = fmaxf(sqrtf(ssq[b * CC + head * DHEAD + e]), 1e-12f);
    float rs = rescale[head];
    const float* g = gram + (size_t)bh * 1024;
    float* a = A + (size_t)bh * 1024;
    for (int d = 0; d < 32; ++d) {
        float kn = fmaxf(sqrtf(ssk[b * CC + head * DHEAD + d]), 1e-12f);
        float v = g[d * 32 + e] / (kn * qn) * rs;
        float m = v;
        for (int off = 16; off > 0; off >>= 1) m = fmaxf(m, __shfl_xor(m, off, 32));
        float ex = __expf(v - m);
        float sum = ex;
        for (int off = 16; off > 0; off >>= 1) sum += __shfl_xor(sum, off, 32);
        a[d * 32 + e] = ex / sum;
    }
}

// =====================================================================
// k5: out[b,h,d,n] = sum_e A[d,e] * v[b,h,e,n]; V tile staged into LDS with
//     async global->LDS DMA (ASYNCcnt path), A broadcast from LDS.
// =====================================================================
__global__ __launch_bounds__(256) void k5_av(const float* __restrict__ A,
                                             const float* __restrict__ vout,
                                             bf16_t* __restrict__ aout) {
    const int NCH = NN / 256;
    int bx = blockIdx.x;
    int chunk = bx % NCH;
    int head = (bx / NCH) % NHEADS;
    int b = bx / (NCH * NHEADS);
    int tid = threadIdx.x;
    int n0 = chunk * 256;
    __shared__ float As[32][33];
    __shared__ float Vs[32][257];
    const float* vp = vout + ((size_t)b * CC + head * DHEAD) * NN + n0;
    // async DMA copy of the 32x256 V tile straight into LDS (b32 per lane)
#pragma unroll
    for (int e = 0; e < 32; ++e) {
        unsigned lds = (unsigned)(unsigned long long)&Vs[e][tid];
        const float* g = vp + (size_t)e * NN + tid;
        asm volatile("global_load_async_to_lds_b32 %0, %1, off"
                     :: "v"(lds), "v"(g) : "memory");
    }
    const float* a = A + (size_t)(b * NHEADS + head) * 1024;
    for (int e = tid; e < 1024; e += 256) As[e >> 5][e & 31] = a[e];
    asm volatile("s_wait_asynccnt 0x0" ::: "memory");
    __syncthreads();
    bf16_t* op = aout + ((size_t)b * CC + head * DHEAD) * NN + n0 + tid;
#pragma unroll 4
    for (int d = 0; d < 32; ++d) {
        float acc = 0.f;
#pragma unroll
        for (int e = 0; e < 32; ++e) acc += As[d][e] * Vs[e][tid];
        op[(size_t)d * NN] = f2bf(acc);
    }
}

// =====================================================================
// k6: 3x3 conv 256->256 as implicit GEMM over K = 9*256, bf16 WMMA.
//     128M x 64N block tile, 8 waves each doing a 32x32 tile (4 WMMA/K-step).
//     Input (67MB bf16) is L2-resident so the 9x tap re-reads are cheap.
// =====================================================================
__global__ __launch_bounds__(256) void k6_conv(const bf16_t* __restrict__ xin,
                                               const float* __restrict__ wproj,
                                               float* __restrict__ out) {
    __shared__ bf16_t Xs[128 * XSTR];    // [n 128][ci 32 permuted]
    __shared__ bf16_t Ws[64 * XSTR];     // [co 64][ci 32 permuted]
    int m0 = blockIdx.x * 128;
    int co0 = blockIdx.y * 64;
    int b = m0 / NN, nbase = m0 % NN;
    int y = nbase >> 8, x0 = nbase & 255;   // x0 in {0,128}
    int tid = threadIdx.x, wave = tid >> 5, lane = tid & 31;
    int mrow = (wave & 3) * 32;          // 32 m's per wave
    int ncol = (wave >> 2) * 32;         // 32 co's per wave
    v8f acc00 = {}, acc01 = {}, acc10 = {}, acc11 = {};
    for (int kk = 0; kk < 9; ++kk) {
        int ky = kk / 3 - 1, kx = kk % 3 - 1;
        int yy = y + ky;
        for (int ci0 = 0; ci0 < CC; ci0 += 32) {
            {   // X tile: thread owns 16 ci's of one spatial point -> two b128 stores
                int e0 = tid * 16;
                int nl = e0 >> 5, k0 = e0 & 31;         // k0 in {0,16}
                int xx = x0 + nl + kx;
                bool ok = (yy >= 0) && (yy < HH) && (xx >= 0) && (xx < WW);
                const bf16_t* src = xin + ((size_t)b * CC + ci0 + k0) * NN + yy * WW + xx;
                bf16_t z = f2bf(0.f);
                v8bf t0, t1;
#pragma unroll
                for (int j = 0; j < 8; ++j) {
                    t0[j] = ok ? src[(size_t)j * NN] : z;
                    t1[j] = ok ? src[(size_t)(j + 8) * NN] : z;
                }
                *(v8bf*)&Xs[nl * XSTR + permk(k0)]     = t0;
                *(v8bf*)&Xs[nl * XSTR + permk(k0 + 8)] = t1;
                // prefetch next ci-slice of the same spatial point
                if (ci0 + 32 < CC && ok)
                    __builtin_prefetch(src + (size_t)32 * NN, 0, 3);
            }
            {   // W tile
                int e0 = tid * 8;
                int co = e0 >> 5, k0 = e0 & 31;
                const float* src = wproj + ((size_t)(co0 + co) * CC + ci0 + k0) * 9 + kk;
                v8bf t;
#pragma unroll
                for (int j = 0; j < 8; ++j) t[j] = f2bf(src[j * 9]);
                *(v8bf*)&Ws[co * XSTR + permk(k0)] = t;
            }
            __syncthreads();
            v16bf a0 = frag_lds(&Xs[mrow * XSTR], XSTR, lane);
            v16bf a1 = frag_lds(&Xs[(mrow + 16) * XSTR], XSTR, lane);
            v16bf b0 = frag_lds(&Ws[ncol * XSTR], XSTR, lane);
            v16bf b1 = frag_lds(&Ws[(ncol + 16) * XSTR], XSTR, lane);
            acc00 = WMMA_BF16(a0, b0, acc00);
            acc01 = WMMA_BF16(a0, b1, acc01);
            acc10 = WMMA_BF16(a1, b0, acc10);
            acc11 = WMMA_BF16(a1, b1, acc11);
            __syncthreads();
        }
    }
    int lhalf = lane >> 4, nlane = lane & 15;
#pragma unroll
    for (int r = 0; r < 8; ++r) {
        int nA = nbase + mrow + r + lhalf * 8;
        int nB = nA + 16;
        int co = co0 + ncol + nlane;
        float* oA = out + ((size_t)b * NN + nA) * CC + co;
        float* oB = out + ((size_t)b * NN + nB) * CC + co;
        oA[0]  = acc00[r];   // NHWC f32 output
        oA[16] = acc01[r];
        oB[0]  = acc10[r];
        oB[16] = acc11[r];
    }
}

extern "C" void kernel_launch(void* const* d_in, const int* in_sizes, int n_in,
                              void* d_out, int out_size, void* d_ws, size_t ws_size,
                              hipStream_t stream) {
    (void)in_sizes; (void)n_in; (void)out_size; (void)ws_size;
    const float* x_in    = (const float*)d_in[0];
    const float* w_qkv   = (const float*)d_in[1];
    const float* w_dw    = (const float*)d_in[2];
    const float* rescale = (const float*)d_in[3];
    const float* w_proj  = (const float*)d_in[4];

    float* out_c = (float*)d_out;
    float* v_out = out_c + (size_t)BATCH * NN * CC;   // second output, NCHW

    char* ws = (char*)d_ws;
    const size_t qkv_bytes = (size_t)BATCH * CQKV * NN * 2;   // bf16
    const size_t qk_bytes  = (size_t)BATCH * CC * NN * 2;     // bf16
    bf16_t* qkv = (bf16_t*)ws;
    bf16_t* qb  = (bf16_t*)(ws + qkv_bytes);
    bf16_t* kb  = (bf16_t*)(ws + qkv_bytes + qk_bytes);
    char* small = ws + qkv_bytes + 2 * qk_bytes;
    float* gram = (float*)small;                               // 16*1024 f32
    float* ssq  = gram + BATCH * NHEADS * 1024;                // 512 f32
    float* ssk  = ssq + BATCH * CC;                            // 512 f32
    float* Amat = ssk + BATCH * CC;                            // 16*1024 f32
    bf16_t* aout = qkv;  // qkv region is dead after k2/k3 -> reuse for conv input

    hipMemsetAsync(gram, 0,
                   (size_t)(BATCH * NHEADS * 1024 + 2 * BATCH * CC) * sizeof(float),
                   stream);

    dim3 g1(BATCH * NN / 64, CQKV / 64);
    k1_qkv<<<g1, 256, 0, stream>>>(x_in, w_qkv, qkv);

    k2_dw<<<BATCH * CQKV * (NN / 256), 256, 0, stream>>>(qkv, w_dw, qb, kb, v_out,
                                                         ssq, ssk);

    k3_gram<<<BATCH * NHEADS * (NN / 2048), 256, 0, stream>>>(qb, kb, gram);

    k4_soft<<<BATCH * NHEADS, 32, 0, stream>>>(gram, ssq, ssk, rescale, Amat);

    k5_av<<<BATCH * NHEADS * (NN / 256), 256, 0, stream>>>(Amat, v_out, aout);

    dim3 g6(BATCH * NN / 128, CC / 64);
    k6_conv<<<g6, 256, 0, stream>>>(aout, w_proj, out_c);
}
